// NerfactoField_6038724018405
// MI455X (gfx1250) — matrix operand order repaired and checked by reading, not compiled, not run on hardware
//
#include <hip/hip_runtime.h>
#include <hip/hip_bf16.h>

typedef __attribute__((ext_vector_type(16))) _Float16 v16h;
typedef __attribute__((ext_vector_type(8)))  _Float16 v8h;
typedef __attribute__((ext_vector_type(2)))  _Float16 v2h;
typedef __attribute__((ext_vector_type(8)))  float    v8f;

#define NUM_LEVELS 16
#define HMOD_MASK  4095u          // HASH_MOD = 4096
#define N_SAMPLES  (4096 * 128)   // R_RAYS * S_SAMP

__constant__ int c_res[NUM_LEVELS] = {16, 22, 30, 42, 58, 80, 111, 154,
                                      212, 294, 406, 561, 776, 1072, 1482, 2048};

// --------------------------------------------------------------------------
// B-fragment tile map (each tile = 32 lanes x 16 halfs = 512 halfs, contiguous
// per lane so a lane's fragment is one 32B load):
//   tiles  0..3  : dw1 (K=32, n0 = tile*16)
//   tiles  4..5  : dw2 (k0 = (tile-4)*32, N=16)
//   tiles  6..13 : cw1 (K-permuted app|geo|sh, row63 zero; j=(t-6)>>1, k0=((t-6)&1)*32)
//   tiles 14..21 : cw2 (j=(t-14)>>1, k0=((t-14)&1)*32)
//   tiles 22..23 : cw3 (k0=(tile-22)*32, cols 3..15 zero)
#define NTILES 24

// ---- WMMA fragment helpers (CDNA5 16x16x32 f16 layouts, wave32) ----------

// A (16x32 f16) from row-major f16 buffer, leading dim 64 halfs.
// lane<16: row=lane, elems 0..7 = K k0+0..7, 8..15 = K k0+16..23
// lane>=16: row=lane-16, elems 0..7 = K k0+8..15, 8..15 = K k0+24..31
// Both 8-half runs are contiguous & 16B aligned -> 2x ds_load_b128.
__device__ inline v16h loadA64(const _Float16* buf, int lane, int k0) {
    int row = lane & 15;
    int kb  = k0 + ((lane & 16) ? 8 : 0);
    const v8h* p = (const v8h*)(buf + row * 64 + kb);
    v8h lo = p[0];
    v8h hi = p[2];      // +16 halfs
    return __builtin_shufflevector(lo, hi, 0, 1, 2, 3, 4, 5, 6, 7,
                                   8, 9, 10, 11, 12, 13, 14, 15);
}

// B (32x16 f16) from pre-swizzled fragment store: one 32B contiguous load.
__device__ inline v16h loadBfrag(const _Float16* frag, int tile, int lane) {
    return *(const v16h*)(frag + (tile * 32 + lane) * 16);
}

__device__ inline v8f wmma16(v16h a, v16h b, v8f c) {
    return __builtin_amdgcn_wmma_f32_16x16x32_f16(
        false, a, false, b, (short)0, c, false, false);
}

// C/D 16x16 f32: vgpr r holds (row r, col=lane) for lane<16, (row r+8, col-16).
// Add bias, ReLU, f16-convert, store into row-major [16][64] LDS tile at n0.
__device__ inline void storeActF16(_Float16* dst, v8f c, int lane, int n0,
                                   const float* bias) {
    int col  = lane & 15;
    int rofs = (lane & 16) ? 8 : 0;
    float b  = bias[n0 + col];
#pragma unroll
    for (int r = 0; r < 8; ++r) {
        float v = c[r] + b;
        dst[(r + rofs) * 64 + n0 + col] = (_Float16)(v > 0.f ? v : 0.f);
    }
}

// --------------------------------------------------------------------------

__global__ __launch_bounds__(256)
void nerfacto_fused(const float* __restrict__ ray_samples,
                    const float* __restrict__ directions,
                    const int*   __restrict__ camera_indices,
                    const float* __restrict__ hash_tables,
                    const float* __restrict__ dw1, const float* __restrict__ db1,
                    const float* __restrict__ dw2, const float* __restrict__ db2,
                    const float* __restrict__ cw1, const float* __restrict__ cb1,
                    const float* __restrict__ cw2, const float* __restrict__ cb2,
                    const float* __restrict__ cw3, const float* __restrict__ cb3,
                    const float* __restrict__ app_embed,
                    float* __restrict__ out) {
    __shared__ __align__(16) _Float16 sBfrag[NTILES * 512];  // 24 KB swizzled weights
    __shared__ float sB1[64], sB2[16], sKb1[64], sKb2[64], sKb3[16];
    __shared__ __align__(16) _Float16 sBuf0[8][16 * 64];     // per-wave ping
    __shared__ __align__(16) _Float16 sBuf1[8][16 * 64];     // per-wave pong

    const int t = threadIdx.x;

    // ---- stage weights as pre-swizzled B fragments (vector stores) ----
    for (int i = t; i < NTILES * 32; i += 256) {   // one 16-half run per iter
        int tile  = i >> 5;
        int lane  = i & 31;
        int laneN = lane & 15;
        int kHi   = (lane & 16) ? 8 : 0;
        v16h v;
        if (tile < 4) {                            // dw1 [32][64]
            int col = tile * 16 + laneN;
#pragma unroll
            for (int e = 0; e < 16; ++e)
                v[e] = (_Float16)dw1[(kHi + (e & 7) + ((e & 8) << 1)) * 64 + col];
        } else if (tile < 6) {                     // dw2 [64][16]
            int k0 = (tile - 4) * 32;
#pragma unroll
            for (int e = 0; e < 16; ++e)
                v[e] = (_Float16)dw2[(k0 + kHi + (e & 7) + ((e & 8) << 1)) * 16 + laneN];
        } else if (tile < 14) {                    // cw1 [63][64], K-permuted
            int tt = tile - 6;
            int col = (tt >> 1) * 16 + laneN;
            int k0  = (tt & 1) * 32;
#pragma unroll
            for (int e = 0; e < 16; ++e) {
                int kk  = k0 + kHi + (e & 7) + ((e & 8) << 1);            // staged row: app|geo|sh|pad
                int src = (kk < 32) ? (kk + 31) : (kk < 63 ? kk - 32 : -1);
                v[e] = (src >= 0) ? (_Float16)cw1[src * 64 + col] : (_Float16)0.f;
            }
        } else if (tile < 22) {                    // cw2 [64][64]
            int tt = tile - 14;
            int col = (tt >> 1) * 16 + laneN;
            int k0  = (tt & 1) * 32;
#pragma unroll
            for (int e = 0; e < 16; ++e)
                v[e] = (_Float16)cw2[(k0 + kHi + (e & 7) + ((e & 8) << 1)) * 64 + col];
        } else {                                   // cw3 [64][3] pad N->16
            int k0 = (tile - 22) * 32;
#pragma unroll
            for (int e = 0; e < 16; ++e)
                v[e] = (laneN < 3) ? (_Float16)cw3[(k0 + kHi + (e & 7) + ((e & 8) << 1)) * 3 + laneN]
                                   : (_Float16)0.f;
        }
        *(v16h*)(sBfrag + i * 16) = v;
    }
    if (t < 64) { sB1[t] = db1[t]; sKb1[t] = cb1[t]; sKb2[t] = cb2[t]; }
    if (t < 16) { sB2[t] = db2[t]; sKb3[t] = (t < 3) ? cb3[t] : 0.f; }
    __syncthreads();

    const int lane = t & 31;
    const int wv   = t >> 5;
    const int m    = lane & 15;       // sample slot in wave's 16-tile
    const int half = lane >> 4;       // work-split bit
    const int base = blockIdx.x * 128 + wv * 16;
    const int gs   = base + m;        // this lane's sample

    _Float16* buf0 = sBuf0[wv];
    _Float16* buf1 = sBuf1[wv];

    float* out_rgb  = out;
    float* out_dens = out + (size_t)N_SAMPLES * 3;
    float* out_geo  = out + (size_t)N_SAMPLES * 4;

    // ---- phase A: hash-grid encode -> buf0[m][0..31] ------------------
    float px, py, pz;
    {
        float x = ray_samples[gs * 3 + 0];
        float y = ray_samples[gs * 3 + 1];
        float z = ray_samples[gs * 3 + 2];
        px = fminf(fmaxf((x + 1.f) * 0.5f, 0.f), 1.f);
        py = fminf(fmaxf((y + 1.f) * 0.5f, 0.f), 1.f);
        pz = fminf(fmaxf((z + 1.f) * 0.5f, 0.f), 1.f);
    }
#pragma unroll
    for (int li = 0; li < 8; ++li) {
        int lvl = half * 8 + li;                 // 8 levels per half-wave
        float rm1 = (float)(c_res[lvl] - 1);
        unsigned cx = (unsigned)fminf(fmaxf(floorf(px * rm1), 0.f), rm1);
        unsigned cy = (unsigned)fminf(fmaxf(floorf(py * rm1), 0.f), rm1);
        unsigned cz = (unsigned)fminf(fmaxf(floorf(pz * rm1), 0.f), rm1);
        unsigned h  = (cx * 1u + cy * 2654435761u + cz * 805459861u) & HMOD_MASK;
        const float* ft = hash_tables + ((size_t)lvl * 4096 + h) * 2;
        float f0 = ft[0], f1 = ft[1];
        v2h hv;
        hv[0] = (_Float16)f0;
        hv[1] = (_Float16)f1;
        *(v2h*)(buf0 + m * 64 + lvl * 2) = hv;   // one ds_store_b32
    }
    __syncthreads();

    // ---- phase B: density layer 1: [16x32] @ [32x64] -> relu -> buf1 --
    {
        v16h a = loadA64(buf0, lane, 0);
#pragma unroll
        for (int j = 0; j < 4; ++j) {
            v8f c = {};
            c = wmma16(a, loadBfrag(sBfrag, j, lane), c);
            storeActF16(buf1, c, lane, j * 16, sB1);
        }
    }
    __syncthreads();

    // ---- phase C: density layer 2: [16x64] @ [64x16] ------------------
    {
        v16h a0 = loadA64(buf1, lane, 0);
        v16h a1 = loadA64(buf1, lane, 32);
        v8f c = {};
        c = wmma16(a0, loadBfrag(sBfrag, 4, lane), c);
        c = wmma16(a1, loadBfrag(sBfrag, 5, lane), c);
        int col  = lane & 15;
        int rofs = (lane & 16) ? 8 : 0;
        float b  = sB2[col];
#pragma unroll
        for (int r = 0; r < 8; ++r) {
            int   sm = r + rofs;
            int   g2 = base + sm;
            float v  = c[r] + b;
            if (col == 0) {
                out_dens[g2] = fmaxf(v, 0.f);                  // density
            } else {
                out_geo[(size_t)g2 * 15 + (col - 1)] = v;      // geo output
                buf0[sm * 64 + 32 + (col - 1)] = (_Float16)v;  // colorIn geo
            }
        }
    }

    // ---- appearance embedding -> colorIn[0..31] (one v16h per half) ---
    {
        int ray = gs >> 7;                       // sample / 128
        int cam = camera_indices[ray];
        const float* ap = app_embed + (size_t)cam * 32 + half * 16;
        v16h av;
#pragma unroll
        for (int j = 0; j < 16; ++j) av[j] = (_Float16)ap[j];
        *(v16h*)(buf0 + m * 64 + half * 16) = av;  // 32B aligned -> 2x b128
    }

    // ---- SH encode -> colorIn[47..62] (8 bands per half-wave) ---------
    {
        float x = directions[gs * 3 + 0];
        float y = directions[gs * 3 + 1];
        float z = directions[gs * 3 + 2];
        float xx = x * x, yy = y * y, zz = z * z;
        float bands[16];
        bands[0]  = 0.28209479177387814f;
        bands[1]  = 0.4886025119029199f * y;
        bands[2]  = 0.4886025119029199f * z;
        bands[3]  = 0.4886025119029199f * x;
        bands[4]  = 1.0925484305920792f * x * y;
        bands[5]  = 1.0925484305920792f * y * z;
        bands[6]  = 0.9461746957575601f * (2.f * zz - xx - yy);
        bands[7]  = 1.0925484305920792f * z * x;
        bands[8]  = 0.5462742152960396f * (xx - yy);
        bands[9]  = 0.5900435899266435f * y * (3.f * xx - yy);
        bands[10] = 2.890611442640554f * x * y * z;
        bands[11] = 0.4570457994644658f * y * (5.f * zz - 1.f);
        bands[12] = 0.3731763325901154f * z * (5.f * zz - 3.f);
        bands[13] = 0.4570457994644658f * x * (5.f * zz - 1.f);
        bands[14] = 1.445305721320277f * z * (xx - yy);
        bands[15] = 0.5900435899266435f * x * (xx - 3.f * yy);
#pragma unroll
        for (int j = 0; j < 8; ++j) {
            int bidx = half * 8 + j;
            buf0[m * 64 + 47 + bidx] = (_Float16)bands[bidx];
        }
        if (half == 0) buf0[m * 64 + 63] = (_Float16)0.f;  // pad col
    }
    __syncthreads();

    // ---- phase D: color layer 1: [16x64] @ [64x64] -> relu -> buf1 ----
    {
        v16h a0 = loadA64(buf0, lane, 0);
        v16h a1 = loadA64(buf0, lane, 32);
#pragma unroll
        for (int j = 0; j < 4; ++j) {
            v8f c = {};
            c = wmma16(a0, loadBfrag(sBfrag, 6 + j * 2 + 0, lane), c);
            c = wmma16(a1, loadBfrag(sBfrag, 6 + j * 2 + 1, lane), c);
            storeActF16(buf1, c, lane, j * 16, sKb1);
        }
    }
    __syncthreads();

    // ---- phase E: color layer 2: [16x64] @ [64x64] -> relu -> buf0 ----
    {
        v16h a0 = loadA64(buf1, lane, 0);
        v16h a1 = loadA64(buf1, lane, 32);
#pragma unroll
        for (int j = 0; j < 4; ++j) {
            v8f c = {};
            c = wmma16(a0, loadBfrag(sBfrag, 14 + j * 2 + 0, lane), c);
            c = wmma16(a1, loadBfrag(sBfrag, 14 + j * 2 + 1, lane), c);
            storeActF16(buf0, c, lane, j * 16, sKb2);
        }
    }
    __syncthreads();

    // ---- phase F: color layer 3: [16x64] @ [64x16(3)] -> sigmoid ------
    {
        v16h a0 = loadA64(buf0, lane, 0);
        v16h a1 = loadA64(buf0, lane, 32);
        v8f c = {};
        c = wmma16(a0, loadBfrag(sBfrag, 22, lane), c);
        c = wmma16(a1, loadBfrag(sBfrag, 23, lane), c);
        int col  = lane & 15;
        int rofs = (lane & 16) ? 8 : 0;
        if (col < 3) {
            float b = sKb3[col];
#pragma unroll
            for (int r = 0; r < 8; ++r) {
                float v = c[r] + b;
                float s = 1.f / (1.f + __expf(-v));
                out_rgb[(size_t)(base + r + rofs) * 3 + col] = s;
            }
        }
    }
}

extern "C" void kernel_launch(void* const* d_in, const int* in_sizes, int n_in,
                              void* d_out, int out_size, void* d_ws, size_t ws_size,
                              hipStream_t stream) {
    (void)in_sizes; (void)n_in; (void)out_size; (void)d_ws; (void)ws_size;
    const float* ray_samples = (const float*)d_in[0];
    const float* directions  = (const float*)d_in[1];
    const int*   cam_idx     = (const int*)d_in[2];
    const float* hash_tables = (const float*)d_in[3];
    const float* dw1 = (const float*)d_in[4];
    const float* db1 = (const float*)d_in[5];
    const float* dw2 = (const float*)d_in[6];
    const float* db2 = (const float*)d_in[7];
    const float* cw1 = (const float*)d_in[8];
    const float* cb1 = (const float*)d_in[9];
    const float* cw2 = (const float*)d_in[10];
    const float* cb2 = (const float*)d_in[11];
    const float* cw3 = (const float*)d_in[12];
    const float* cb3 = (const float*)d_in[13];
    const float* app = (const float*)d_in[14];
    float* out = (float*)d_out;

    dim3 grid(N_SAMPLES / 128);   // 4096 blocks, each: 8 waves x 16 samples
    dim3 block(256);
    nerfacto_fused<<<grid, block, 0, stream>>>(
        ray_samples, directions, cam_idx, hash_tables,
        dw1, db1, dw2, db2, cw1, cb1, cw2, cb2, cw3, cb3, app, out);
}
